// JCDD_G_layer_23794118820499
// MI455X (gfx1250) — matrix operand order repaired and checked by reading, not compiled, not run on hardware
//
#include <hip/hip_runtime.h>
#include <hip/hip_bf16.h>

typedef __bf16 bf16;
typedef bf16  bf16x4  __attribute__((ext_vector_type(4)));
typedef bf16  bf16x8  __attribute__((ext_vector_type(8)));
typedef bf16  bf16x16 __attribute__((ext_vector_type(16)));
typedef float floatx8 __attribute__((ext_vector_type(8)));

union Frag { bf16x8 h[2]; bf16x16 v; };

#define LDPCN  4096
#define MA_    12288
static constexpr float SQ2     = 1.41421356237309515f;
static constexpr float INV_SQ2 = 0.70710678118654752f;

// ---------------------------------------------------------------------------
// CDNA5 async memory->LDS copy (ASYNCcnt-tracked), per ISA ch.10.7/15.18
// ---------------------------------------------------------------------------
__device__ __forceinline__ void async_copy_b128(unsigned lds_off, const void* g) {
    asm volatile("global_load_async_to_lds_b128 %0, %1, off"
                 :: "v"(lds_off), "v"((unsigned long long)(size_t)g)
                 : "memory");
}
__device__ __forceinline__ void wait_async0() {
    asm volatile("s_wait_asynccnt 0x0" ::: "memory");
}

// ---------------------------------------------------------------------------
// A (fp32, 12288x4096) -> bf16 row-major + bf16 transposed, tiled 64x64 so
// both layouts are written fully coalesced (b64 row-major, b128 transposed stores).
// grid = (4096/64, 12288/64)
// ---------------------------------------------------------------------------
__global__ __launch_bounds__(256) void k_convertA(const float* __restrict__ A,
                                                  bf16* __restrict__ Abf,
                                                  bf16* __restrict__ ATbf) {
    __shared__ bf16 tile[64][80];           // [n_local][m_local], padded
    const int n0 = blockIdx.x * 64;         // column block in A
    const int m0 = blockIdx.y * 64;         // row block in A
    const int tid = threadIdx.x;

    #pragma unroll
    for (int i = 0; i < 4; ++i) {
        const int idx = tid + i * 256;          // 1024 chunks of 4 floats
        const int r = idx >> 4;                 // 0..63
        const int c = (idx & 15) * 4;           // 0..60
        const float4 v = *(const float4*)(A + (size_t)(m0 + r) * LDPCN + n0 + c);
        bf16x4 h = { (bf16)v.x, (bf16)v.y, (bf16)v.z, (bf16)v.w };
        *(bf16x4*)(Abf + (size_t)(m0 + r) * LDPCN + n0 + c) = h;
        tile[c + 0][r] = h[0];
        tile[c + 1][r] = h[1];
        tile[c + 2][r] = h[2];
        tile[c + 3][r] = h[3];
    }
    __syncthreads();
    #pragma unroll
    for (int i = 0; i < 2; ++i) {
        const int idx = tid + i * 256;          // 512 chunks of 8 bf16
        const int r = idx >> 3;                 // 0..63 (n_local)
        const int c = (idx & 7) * 8;            // 0..56 (m_local)
        *(bf16x8*)(ATbf + (size_t)(n0 + r) * MA_ + m0 + c) =
            *(const bf16x8*)&tile[r][c];
    }
}

// ---------------------------------------------------------------------------
// r^T[b,m] = theta[m] - z[b,m] - lambda1[b,m]  (bf16, batch-major), x4 vector
// ---------------------------------------------------------------------------
__global__ __launch_bounds__(256) void k_prep_r(const float* __restrict__ theta,
                                                const float* __restrict__ z,
                                                const float* __restrict__ l1,
                                                bf16* __restrict__ rbf) {
    const size_t i4 = ((size_t)blockIdx.x * 256 + threadIdx.x) * 4;
    const int m = (int)(i4 % MA_);
    const float4 zv = *(const float4*)(z + i4);
    const float4 lv = *(const float4*)(l1 + i4);
    const float4 tv = *(const float4*)(theta + m);
    bf16x4 o = { (bf16)(tv.x - zv.x - lv.x), (bf16)(tv.y - zv.y - lv.y),
                 (bf16)(tv.z - zv.z - lv.z), (bf16)(tv.w - zv.w - lv.w) };
    *(bf16x4*)(rbf + i4) = o;
}

// ---------------------------------------------------------------------------
// GEMM: C(512 x N) = X(512 x K) @ W(K x N), bf16 in, fp32 acc.
// B operand supplied as Wt = W^T (N x K row-major), so BOTH operands stage
// with contiguous b128 async copies and frag-load with contiguous b128 ds
// reads (per the CDNA5 16-bit A/B lane layouts). Double-buffered LDS tiles,
// GLOBAL_LOAD_ASYNC_TO_LDS_B128 staging overlapped with WMMA.
//   EPI==1: u-update epilogue   (X=r^T,  Wt=ATbf, N=4096,  K=12288)
//   EPI==2: z/lambda1 epilogue  (X=u^T,  Wt=Abf,  N=12288, K=4096)
// Workgroup tile 64(M) x 128(N), 8 waves of 32x32, KT=32.
// ---------------------------------------------------------------------------
template <int EPI>
__global__ __launch_bounds__(256) void k_gemm(
    const bf16* __restrict__ Xg, const bf16* __restrict__ Wt,
    int K, int N,
    const float* __restrict__ lambda_W, const float* __restrict__ F,
    const float* __restrict__ LamA,
    const float* __restrict__ theta, const float* __restrict__ z_old,
    const float* __restrict__ lam_old,
    const float* __restrict__ s_miu, const float* __restrict__ s_alpha,
    const float* __restrict__ s_relax, const float* __restrict__ s_acc,
    float* __restrict__ out0, float* __restrict__ out1, bf16* __restrict__ ubf) {

    __shared__ bf16 xs [2][64][40];     // X tile  (M x K), padded rows (80 B)
    __shared__ bf16 wsn[2][128][40];    // Wt tile (N x K), padded rows

    const int tid  = threadIdx.x;
    const int lane = tid & 31;
    const int wave = tid >> 5;
    const int wm = wave >> 2;           // 0..1
    const int wn = wave & 3;            // 0..3
    const int lrow = lane & 15;
    const int hi   = lane >> 4;         // 0 or 1
    const int m0 = blockIdx.y * 64;
    const int n0 = blockIdx.x * 128;

    const int xrow = tid >> 2, xch = (tid & 3) * 8;
    const int T = K >> 5;

    floatx8 acc[2][2] = {};

    auto issue = [&](int kt, int d) {
        const int k0 = kt << 5;
        async_copy_b128((unsigned)(size_t)&xs[d][xrow][xch],
                        Xg + (size_t)(m0 + xrow) * K + k0 + xch);
        #pragma unroll
        for (int i = 0; i < 2; ++i) {
            const int idx = tid + i * 256;
            const int row = idx >> 2, ch = (idx & 3) * 8;
            async_copy_b128((unsigned)(size_t)&wsn[d][row][ch],
                            Wt + (size_t)(n0 + row) * K + k0 + ch);
        }
        if (kt + 2 < T)   // deepen the pipeline past the 1-deep async buffer
            __builtin_prefetch(Wt + (size_t)(n0 + (tid >> 1)) * K + k0 + 64, 0, 0);
    };

    issue(0, 0);

    for (int kt = 0; kt < T; ++kt) {
        const int cur = kt & 1, nxt = cur ^ 1;
        wait_async0();                  // tile kt resident in LDS
        __syncthreads();                // visible to all waves; nxt buffer free
        if (kt + 1 < T) issue(kt + 1, nxt);

        Frag a[2], b[2];
        #pragma unroll
        for (int mi = 0; mi < 2; ++mi) {
            const bf16x8* xr = (const bf16x8*)&xs[cur][wm * 32 + mi * 16 + lrow][0];
            a[mi].h[0] = xr[hi];        // K 0..7   / 8..15
            a[mi].h[1] = xr[2 + hi];    // K 16..23 / 24..31
        }
        #pragma unroll
        for (int ni = 0; ni < 2; ++ni) {
            const bf16x8* wr = (const bf16x8*)&wsn[cur][wn * 32 + ni * 16 + lrow][0];
            b[ni].h[0] = wr[hi];         // K 0..7   / 8..15
            b[ni].h[1] = wr[2 + hi];     // K 16..23 / 24..31
        }
        #pragma unroll
        for (int mi = 0; mi < 2; ++mi)
            #pragma unroll
            for (int ni = 0; ni < 2; ++ni)
                acc[mi][ni] = __builtin_amdgcn_wmma_f32_16x16x32_bf16(
                    false, a[mi].v, false, b[ni].v, (short)0, acc[mi][ni],
                    false, false);
    }

    // ---- fused epilogues ---------------------------------------------------
    if constexpr (EPI == 1) {
        const float miu = *s_miu, alpha = *s_alpha;
        #pragma unroll
        for (int ni = 0; ni < 2; ++ni) {
            const int i  = n0 + wn * 32 + ni * 16 + lrow;  // 0..4095
            const int l  = i >> 4;
            const int t  = (i >> 1) & 7;
            const int qq = i & 1;
            const int c  = qq * 8 + t;
            const float la = LamA[i];
            #pragma unroll
            for (int mi = 0; mi < 2; ++mi) {
                #pragma unroll
                for (int r = 0; r < 8; ++r) {
                    const int bb = m0 + wm * 32 + mi * 16 + hi * 8 + r;
                    const float lw    = lambda_W[bb];
                    const float res   = F[(size_t)bb * (16 * 256) + c * 256 + l];
                    const float fbres = 2.f * lw - 2.f * SQ2 * res;
                    const float qv    = miu * acc[mi][ni][r] + fbres - alpha;
                    const float wv    = 1.f / (miu * la + 4.f * lw - 2.f * alpha);
                    float uv = qv * wv;
                    uv = fminf(fmaxf(uv, 0.f), 1.f);
                    const size_t off = (size_t)bb * LDPCN + i;
                    out0[off] = uv;
                    ubf[off]  = (bf16)uv;
                }
            }
        }
    } else {
        const float relax = *s_relax, accs = *s_acc;
        #pragma unroll
        for (int ni = 0; ni < 2; ++ni) {
            const int j = n0 + wn * 32 + ni * 16 + lrow;   // 0..12287
            const float th = theta[j];
            #pragma unroll
            for (int mi = 0; mi < 2; ++mi) {
                #pragma unroll
                for (int r = 0; r < 8; ++r) {
                    const int bb = m0 + wm * 32 + mi * 16 + hi * 8 + r;
                    const size_t off = (size_t)bb * MA_ + j;
                    const float zo = z_old[off];
                    const float lo = lam_old[off];
                    const float zt = th - relax * acc[mi][ni][r]
                                   - (1.f - relax) * (th - zo) - lo;
                    const float zv = fmaxf(zt, 0.f);
                    const float lv = fmaxf(-zt, 0.f);
                    out0[off] = zv + accs * (zv - zo);
                    out1[off] = lv + accs * (lv - lo);
                }
            }
        }
    }
}

// ---------------------------------------------------------------------------
// Per-batch MMSE chain: X, XXp, R, inv(R), W, F_new, lambda_W, acc_new
// ---------------------------------------------------------------------------
__global__ __launch_bounds__(256) void k_mmse(
    const float* __restrict__ sigma2I, const float* __restrict__ Y,
    const float* __restrict__ YYp, const float* __restrict__ rXp,
    const float* __restrict__ iXp, const float* __restrict__ lambda_W,
    const float* __restrict__ accnew_in, const float* __restrict__ s_factor,
    const float* __restrict__ u, float* __restrict__ outF,
    float* __restrict__ outLW, float* __restrict__ outACC) {

    __shared__ float XXp[16][544];
    __shared__ float Raug[16][33];
    __shared__ float T[16][8];
    __shared__ float P[16][16];
    __shared__ float G[16][17];

    const int b = blockIdx.x, tid = threadIdx.x;
    const float* ub = u + (size_t)b * LDPCN;

    for (int e = tid; e < 16 * 544; e += 256) {
        const int row = e / 544, col = e % 544;
        const int half = (col < 272) ? 0 : 1;
        const int cc = col - half * 272;
        const int tt = (row < 8) ? row : row - 8;
        float rxv, ixv;
        if (cc < 256) {
            rxv = (1.f - 2.f * ub[cc * 16 + tt * 2 + 0]) * INV_SQ2;
            ixv = (1.f - 2.f * ub[cc * 16 + tt * 2 + 1]) * INV_SQ2;
        } else {
            rxv = rXp[((size_t)b * 8 + tt) * 16 + (cc - 256)];
            ixv = iXp[((size_t)b * 8 + tt) * 16 + (cc - 256)];
        }
        float v;
        if (row < 8) v = half ? ixv : rxv;
        else         v = half ? rxv : -ixv;
        XXp[row][col] = v;
    }
    __syncthreads();

    {
        const int r = tid >> 4, c = tid & 15;
        float s = 0.f;
        for (int k = 0; k < 544; ++k) s += XXp[r][k] * XXp[c][k];
        Raug[r][c]      = s + sigma2I[((size_t)b * 16 + r) * 16 + c];
        Raug[r][16 + c] = (r == c) ? 1.f : 0.f;
    }
    if (tid < 128) {
        const int r = tid >> 3, c = tid & 7;
        float s = 0.f;
        const float* yr = YYp + ((size_t)b * 8 + c) * 544;
        for (int k = 0; k < 544; ++k) s += XXp[r][k] * yr[k];
        T[r][c] = s;
    }

    const int gr = tid >> 5, gc = tid & 31;
    for (int p = 0; p < 16; ++p) {
        __syncthreads();
        const float ipiv = 1.f / Raug[p][p];
        __syncthreads();
        if (tid < 32) Raug[p][tid] *= ipiv;
        __syncthreads();
        const float f0 = Raug[gr][p];
        const float f1 = Raug[gr + 8][p];
        __syncthreads();
        if (gr != p)     Raug[gr][gc]     -= f0 * Raug[p][gc];
        if (gr + 8 != p) Raug[gr + 8][gc] -= f1 * Raug[p][gc];
    }
    __syncthreads();

    if (tid < 128) {
        const int r = tid >> 3, c = tid & 7;
        float s = 0.f;
        for (int k = 0; k < 16; ++k) s += Raug[r][16 + k] * T[k][c];
        if (r < 8) { P[r][c] = s;          P[r + 8][c + 8] = s; }
        else       { P[r - 8][c + 8] = s;  P[r][c] = -s; }
    }
    __syncthreads();

    {
        const int r = tid >> 4, c = tid & 15;
        float s = 0.f;
        for (int k = 0; k < 16; ++k) s += P[r][k] * P[c][k];
        G[r][c] = s;
    }
    __syncthreads();

    const float lwn = (*s_factor) * lambda_W[b];
    for (int e = tid; e < 16 * 256; e += 256) {
        const int row = e >> 8, l = e & 255;
        float s = 0.f;
        for (int k = 0; k < 16; ++k)
            s += P[row][k] * Y[((size_t)b * 16 + k) * 256 + l];
        float gx = 0.f;
        for (int k = 0; k < 16; ++k) {
            const float xv = (k < 8) ? XXp[k][l] : XXp[k - 8][272 + l];
            gx += G[row][k] * xv;
        }
        const float xr = (row < 8) ? XXp[row][l] : XXp[row - 8][272 + l];
        outF[((size_t)b * 16 + row) * 256 + l] = s + lwn * xr - gx;
    }
    if (tid == 0) { outLW[b] = lwn; outACC[b] = accnew_in[b]; }
}

// ---------------------------------------------------------------------------
extern "C" void kernel_launch(void* const* d_in, const int* in_sizes, int n_in,
                              void* d_out, int out_size, void* d_ws, size_t ws_size,
                              hipStream_t stream) {
    const float* sigma2I = (const float*)d_in[0];
    const float* Y       = (const float*)d_in[1];
    const float* YYp     = (const float*)d_in[2];
    const float* rXp     = (const float*)d_in[3];
    const float* iXp     = (const float*)d_in[4];
    const float* lamW    = (const float*)d_in[5];
    const float* F       = (const float*)d_in[6];
    const float* z       = (const float*)d_in[8];
    const float* l1      = (const float*)d_in[9];
    const float* accnew  = (const float*)d_in[10];
    const float* A       = (const float*)d_in[11];
    const float* theta   = (const float*)d_in[12];
    const float* LamA    = (const float*)d_in[13];
    const float* s_miu   = (const float*)d_in[14];
    const float* s_alpha = (const float*)d_in[15];
    const float* s_fact  = (const float*)d_in[16];
    const float* s_relax = (const float*)d_in[17];
    const float* s_acc   = (const float*)d_in[18];

    float* out     = (float*)d_out;
    float* out_lw  = out;                       // 512
    float* out_F   = out_lw + 512;              // 512*16*256
    float* out_u   = out_F + 512 * 4096;        // 512*4096
    float* out_z   = out_u + 512 * 4096;        // 512*12288
    float* out_l1  = out_z + 512 * 12288;       // 512*12288
    float* out_acc = out_l1 + 512 * 12288;      // 512

    char* ws   = (char*)d_ws;
    bf16* Abf  = (bf16*)ws;                                  // 100,663,296 B
    bf16* ATbf = (bf16*)(ws + 100663296ull);                 // 100,663,296 B
    bf16* rbf  = (bf16*)(ws + 201326592ull);                 //  12,582,912 B
    bf16* ubf  = (bf16*)(ws + 213909504ull);                 //   4,194,304 B

    dim3 gc(4096 / 64, 12288 / 64);
    k_convertA<<<gc, 256, 0, stream>>>(A, Abf, ATbf);
    k_prep_r<<<6144, 256, 0, stream>>>(theta, z, l1, rbf);

    // GEMM1: out_u^T = r^T @ A      (B operand = columns of A = rows of ATbf)
    dim3 g1(4096 / 128, 512 / 64);
    k_gemm<1><<<g1, 256, 0, stream>>>(rbf, ATbf, MA_, LDPCN,
        lamW, F, LamA, nullptr, nullptr, nullptr,
        s_miu, s_alpha, s_relax, s_acc, out_u, nullptr, ubf);

    // GEMM2: (A@u)^T = u^T @ A^T    (B operand = columns of A^T = rows of Abf)
    dim3 g2(12288 / 128, 512 / 64);
    k_gemm<2><<<g2, 256, 0, stream>>>(ubf, Abf, LDPCN, MA_,
        lamW, F, LamA, theta, z, l1,
        s_miu, s_alpha, s_relax, s_acc, out_z, out_l1, nullptr);

    k_mmse<<<512, 256, 0, stream>>>(sigma2I, Y, YYp, rXp, iXp, lamW,
        accnew, s_fact, out_u, out_F, out_lw, out_acc);
}
